// Superpixel_Vit_52329881534581
// MI455X (gfx1250) — matrix operand order repaired and checked by reading, not compiled, not run on hardware
//
#include <hip/hip_runtime.h>
#include <math.h>

// ---------------------------------------------------------------------------
// Superpixel-ViT forward for MI455X (gfx1250, wave32, WMMA).
// All GEMMs run through a bf16 WMMA kernel (v_wmma_f32_16x16x32_bf16),
// fp32 everywhere else (graph scatters, layernorms, softmaxes).
// ---------------------------------------------------------------------------

typedef __attribute__((ext_vector_type(16))) __bf16 v16bf;
typedef __attribute__((ext_vector_type(8)))  __bf16 v8bf;
typedef __attribute__((ext_vector_type(8)))  float  v8f;

#define N_PATCH 102400
#define IN_DIM  1024
#define HIDD    512
#define N_SP    6400
#define BATCH   64
#define PP      100
#define SEQ     101
#define EP_N    819200
#define ES_N    51200
#define HEADS   8
#define DHD     64
#define NCLS    2

static inline int cdiv_i(int a, int b) { return (a + b - 1) / b; }

// ---------------------------------------------------------------------------
// Generic fills
// ---------------------------------------------------------------------------
__global__ void k_fill_f(float* p, long long n, float v) {
  long long i = (long long)blockIdx.x * blockDim.x + threadIdx.x;
  long long st = (long long)gridDim.x * blockDim.x;
  for (; i < n; i += st) p[i] = v;
}
__global__ void k_fill_u(unsigned* p, long long n, unsigned v) {
  long long i = (long long)blockIdx.x * blockDim.x + threadIdx.x;
  long long st = (long long)gridDim.x * blockDim.x;
  for (; i < n; i += st) p[i] = v;
}

// ---------------------------------------------------------------------------
// LayerNorm: one block (256 thr = 8 waves) per row; safe in-place.
// ---------------------------------------------------------------------------
__global__ __launch_bounds__(256) void k_ln(const float* __restrict__ x,
                                            const float* __restrict__ g,
                                            const float* __restrict__ b,
                                            float* __restrict__ y,
                                            int D, float eps) {
  int row = blockIdx.x;
  const float* xr = x + (size_t)row * D;
  float* yr = y + (size_t)row * D;
  __shared__ float sh[8];
  __shared__ float smu, srs;
  int tid = threadIdx.x, lane = tid & 31, wv = tid >> 5;

  float s = 0.f;
  for (int i = tid; i < D; i += 256) s += xr[i];
#pragma unroll
  for (int o = 16; o > 0; o >>= 1) s += __shfl_xor(s, o, 32);
  if (lane == 0) sh[wv] = s;
  __syncthreads();
  if (tid == 0) { float t = 0.f; for (int i = 0; i < 8; ++i) t += sh[i]; smu = t / D; }
  __syncthreads();
  float mu = smu;

  float v = 0.f;
  for (int i = tid; i < D; i += 256) { float d = xr[i] - mu; v += d * d; }
#pragma unroll
  for (int o = 16; o > 0; o >>= 1) v += __shfl_xor(v, o, 32);
  __syncthreads();
  if (lane == 0) sh[wv] = v;
  __syncthreads();
  if (tid == 0) { float t = 0.f; for (int i = 0; i < 8; ++i) t += sh[i]; srs = rsqrtf(t / D + eps); }
  __syncthreads();
  float rs = srs;

  for (int i = tid; i < D; i += 256) yr[i] = (xr[i] - mu) * rs * g[i] + b[i];
}

// ---------------------------------------------------------------------------
// bf16 WMMA GEMM:  C[N,M] = act(A[N,K] @ B[K,M] + bias + res)
// Block tile 128x128x32, 8 wave32 waves, each wave: 2x4 WMMA frags.
// Software-pipelined: next tile's global loads (b128, unconditional via
// row clamp) are issued before the current tile's WMMAs. fp32->bf16 packed
// conversion in registers, wide LDS stores (b64 for A, b32 pairs for B).
// act: 0=none 1=relu 2=gelu(tanh)
// ---------------------------------------------------------------------------
__device__ inline float gelu_tanh(float x) {
  const float c0 = 0.7978845608028654f, c1 = 0.044715f;
  float t = tanhf(c0 * (x + c1 * x * x * x));
  return 0.5f * x * (1.f + t);
}

#define BM 128
#define BN 128
#define BK 32

__global__ __launch_bounds__(256) void k_gemm_bf16(
    const float* __restrict__ A, const float* __restrict__ B,
    const float* __restrict__ bias, const float* __restrict__ res,
    float* __restrict__ C, int N, int K, int M, int act) {
  __shared__ __bf16 sA[BM][BK];   // row-major  [m][k]
  __shared__ __bf16 sB[BN][BK];   // transposed [n][k] so B-frag is k-contiguous

  const int m0 = blockIdx.x * BM;
  const int n0 = blockIdx.y * BN;
  const int tid = threadIdx.x;
  const int lane = tid & 31;
  const int wave = tid >> 5;            // 0..7
  const int wm = wave >> 1;             // 0..3 : 32-row band
  const int wn = wave & 1;              // 0..1 : 64-col band
  const int rlo = lane & 15;
  const int khalf = lane >> 4;          // 0 or 1

  // ---- staging register state (pipelined) ----
  float4 ar[4];
  float4 br0[2], br1[2];

  // A: thread covers 4 groups of 4 consecutive K values; rows clamped (no
  // zero-fill needed: OOB A rows only feed OOB C rows, guarded at store).
  const int a_r[4]  = { (tid + 0)   >> 3, (tid + 256) >> 3,
                        (tid + 512) >> 3, (tid + 768) >> 3 };
  const int a_c4[4] = { tid & 7, tid & 7, tid & 7, tid & 7 };
  // B: thread covers 2 groups: 4 consecutive cols x 2 consecutive k rows.
  const int b_n4[2] = { tid & 31, tid & 31 };
  const int b_kk[2] = { tid >> 5, (tid + 256) >> 5 };

  auto load_tiles = [&](int k0) {
#pragma unroll
    for (int p = 0; p < 4; ++p) {
      int gr = m0 + a_r[p];
      gr = (gr < N) ? gr : (N - 1);
      ar[p] = *(const float4*)(A + (size_t)gr * K + (k0 + a_c4[p] * 4));
    }
#pragma unroll
    for (int p = 0; p < 2; ++p) {
      const float* bp = B + (size_t)(k0 + b_kk[p] * 2) * M + (n0 + b_n4[p] * 4);
      br0[p] = *(const float4*)bp;
      br1[p] = *(const float4*)(bp + M);
    }
  };
  auto store_tiles = [&]() {
#pragma unroll
    for (int p = 0; p < 4; ++p) {
      union { uint2 u; __bf16 h[4]; } pk;
      pk.h[0] = (__bf16)ar[p].x; pk.h[1] = (__bf16)ar[p].y;
      pk.h[2] = (__bf16)ar[p].z; pk.h[3] = (__bf16)ar[p].w;
      *(uint2*)&sA[a_r[p]][a_c4[p] * 4] = pk.u;
    }
#pragma unroll
    for (int p = 0; p < 2; ++p) {
      float f0[4] = { br0[p].x, br0[p].y, br0[p].z, br0[p].w };
      float f1[4] = { br1[p].x, br1[p].y, br1[p].z, br1[p].w };
#pragma unroll
      for (int i = 0; i < 4; ++i) {
        union { unsigned u; __bf16 h[2]; } pk;
        pk.h[0] = (__bf16)f0[i]; pk.h[1] = (__bf16)f1[i];
        *(unsigned*)&sB[b_n4[p] * 4 + i][b_kk[p] * 2] = pk.u;
      }
    }
  };

  v8f acc[2][4];
#pragma unroll
  for (int i = 0; i < 2; ++i)
#pragma unroll
    for (int j = 0; j < 4; ++j)
#pragma unroll
      for (int e = 0; e < 8; ++e) acc[i][j][e] = 0.f;

  load_tiles(0);

  for (int k0 = 0; k0 < K; k0 += BK) {
    __syncthreads();               // previous tile fully consumed
    store_tiles();
    __syncthreads();               // tile visible to all waves

    if (k0 + BK < K) load_tiles(k0 + BK);   // overlap with WMMA below
    if (k0 + 2 * BK < K) {
      int gr = m0 + (tid & 127);
      gr = (gr < N) ? gr : (N - 1);
      __builtin_prefetch(A + (size_t)gr * K + (k0 + 2 * BK), 0, 1);
    }

    // A fragments (ISA 7.12.2): lane<16 row=rlo K{0..7,16..23}; lane>=16 K{8..15,24..31}
    v16bf afr[2];
#pragma unroll
    for (int i = 0; i < 2; ++i) {
      const __bf16* pa = &sA[wm * 32 + i * 16 + rlo][0];
      union { v16bf v; v8bf h[2]; } u;
      u.h[0] = *(const v8bf*)(pa + khalf * 8);
      u.h[1] = *(const v8bf*)(pa + 16 + khalf * 8);
      afr[i] = u.v;
    }
    // B fragments: lanes 0-15 cols, K{0..7,16..23} ; lanes 16-31 same cols, K{8..15,24..31}
    v16bf bfr[4];
#pragma unroll
    for (int j = 0; j < 4; ++j) {
      const __bf16* pb = &sB[wn * 64 + j * 16 + rlo][0];
      union { v16bf v; v8bf h[2]; } u;
      u.h[0] = *(const v8bf*)(pb + khalf * 8);
      u.h[1] = *(const v8bf*)(pb + 16 + khalf * 8);
      bfr[j] = u.v;
    }

#pragma unroll
    for (int i = 0; i < 2; ++i)
#pragma unroll
      for (int j = 0; j < 4; ++j)
        acc[i][j] = __builtin_amdgcn_wmma_f32_16x16x32_bf16(
            false, afr[i], false, bfr[j], (short)0, acc[i][j], false, false);
  }

  // epilogue: C/D layout -> VGPR e: lanes 0-15 M=e, lanes 16-31 M=8+e
#pragma unroll
  for (int i = 0; i < 2; ++i)
#pragma unroll
    for (int j = 0; j < 4; ++j)
#pragma unroll
      for (int e = 0; e < 8; ++e) {
        int row = m0 + wm * 32 + i * 16 + khalf * 8 + e;
        int col = n0 + wn * 64 + j * 16 + rlo;
        if (row < N) {
          float v = acc[i][j][e];
          if (bias) v += bias[col];
          if (res)  v += res[(size_t)row * M + col];
          if (act == 1) v = fmaxf(v, 0.f);
          else if (act == 2) v = gelu_tanh(v);
          C[(size_t)row * M + col] = v;
        }
      }
}

// ---------------------------------------------------------------------------
// Graph ops
// ---------------------------------------------------------------------------
__global__ __launch_bounds__(256) void k_scatter_rows(const float* __restrict__ src,
                                                      const int* __restrict__ es,
                                                      const int* __restrict__ ed,
                                                      float* __restrict__ dst,
                                                      int nE, int D) {
  int e = blockIdx.x;
  if (e >= nE) return;
  const float* s = src + (size_t)es[e] * D;
  float* d = dst + (size_t)ed[e] * D;
  for (int i = threadIdx.x; i < D; i += blockDim.x) atomicAdd(&d[i], s[i]);
}

__global__ void k_deg(const int* __restrict__ ed, float* __restrict__ deg, int nE) {
  int e = blockIdx.x * blockDim.x + threadIdx.x;
  if (e < nE) atomicAdd(&deg[ed[e]], 1.f);
}

__global__ __launch_bounds__(256) void k_divdeg(float* __restrict__ buf,
                                                const float* __restrict__ deg, int D) {
  int row = blockIdx.x;
  float d = deg[row];
  float inv = 1.f / fmaxf(d, 1.f);
  float* r = buf + (size_t)row * D;
  for (int i = threadIdx.x; i < D; i += 256) r[i] *= inv;
}

// ---------------------------------------------------------------------------
// Attention-gate pooling (segment softmax over superpixels)
// ---------------------------------------------------------------------------
__global__ __launch_bounds__(256) void k_gate_dot(const float* __restrict__ g1,
                                                  const float* __restrict__ W2,
                                                  const float* __restrict__ b2,
                                                  float* __restrict__ gate, int N) {
  int row = blockIdx.x * 8 + (threadIdx.x >> 5);
  if (row >= N) return;
  int lane = threadIdx.x & 31;
  const float* r = g1 + (size_t)row * 256;
  float s = 0.f;
  for (int i = lane; i < 256; i += 32) s += r[i] * W2[i];
#pragma unroll
  for (int o = 16; o > 0; o >>= 1) s += __shfl_xor(s, o, 32);
  if (lane == 0) gate[row] = s + b2[0];
}

__device__ inline unsigned fenc(float f) {
  unsigned u = __float_as_uint(f);
  return (u & 0x80000000u) ? ~u : (u | 0x80000000u);
}
__device__ inline float fdec(unsigned u) {
  unsigned v = (u & 0x80000000u) ? (u & 0x7FFFFFFFu) : ~u;
  return __uint_as_float(v);
}

__global__ void k_seg_max(const float* __restrict__ gate, const int* __restrict__ spa,
                          unsigned* __restrict__ m, int N) {
  int i = blockIdx.x * blockDim.x + threadIdx.x;
  if (i < N) atomicMax(&m[spa[i]], fenc(gate[i]));
}
__global__ void k_seg_exp(float* __restrict__ gate, const int* __restrict__ spa,
                          const unsigned* __restrict__ m, float* __restrict__ z, int N) {
  int i = blockIdx.x * blockDim.x + threadIdx.x;
  if (i < N) {
    float e = __expf(gate[i] - fdec(m[spa[i]]));
    gate[i] = e;
    atomicAdd(&z[spa[i]], e);
  }
}
__global__ __launch_bounds__(256) void k_pool(const float* __restrict__ h,
                                              const float* __restrict__ gate,
                                              const float* __restrict__ z,
                                              const int* __restrict__ spa,
                                              float* __restrict__ sp, int D) {
  int i = blockIdx.x;
  int s = spa[i];
  float w = gate[i] / z[s];
  const float* hr = h + (size_t)i * D;
  float* o = sp + (size_t)s * D;
  for (int d = threadIdx.x; d < D; d += 256) atomicAdd(&o[d], w * hr[d]);
}

// ---------------------------------------------------------------------------
// Token assembly: t[b,0]=cls, t[b,1+j]=tbody[b*100+j]; all + pos[j]
// ---------------------------------------------------------------------------
__global__ void k_tokens(const float* __restrict__ tb, const float* __restrict__ cls,
                         const float* __restrict__ pos, float* __restrict__ t) {
  long long total = (long long)BATCH * SEQ * HIDD;
  long long idx = (long long)blockIdx.x * blockDim.x + threadIdx.x;
  if (idx >= total) return;
  int d = (int)(idx % HIDD);
  int j = (int)((idx / HIDD) % SEQ);
  int b = (int)(idx / ((long long)HIDD * SEQ));
  float v = (j == 0) ? cls[d] : tb[((size_t)b * PP + (j - 1)) * HIDD + d];
  t[idx] = v + pos[(size_t)j * HIDD + d];
}

// ---------------------------------------------------------------------------
// Multi-head attention: one block per (b,h,q). S=101, DH=64.
// ---------------------------------------------------------------------------
__global__ __launch_bounds__(128) void k_attn(const float* __restrict__ qkv,
                                              float* __restrict__ out) {
  int q = blockIdx.x % SEQ;
  int bh = blockIdx.x / SEQ;
  int h = bh % HEADS;
  int b = bh / HEADS;
  int tid = threadIdx.x;

  __shared__ float sq[DHD];
  __shared__ float sc[SEQ];
  __shared__ float sinv;

  const float* base = qkv + (size_t)b * SEQ * 3 * HIDD;
  if (tid < DHD) sq[tid] = base[(size_t)q * 3 * HIDD + h * DHD + tid];
  __syncthreads();

  for (int k = tid; k < SEQ; k += 128) {
    const float* kr = base + (size_t)k * 3 * HIDD + HIDD + h * DHD;
    float s = 0.f;
#pragma unroll 8
    for (int d = 0; d < DHD; ++d) s += sq[d] * kr[d];
    sc[k] = s * 0.125f;  // 1/sqrt(64)
  }
  __syncthreads();

  if (tid == 0) {
    float mx = -1e30f;
    for (int k = 0; k < SEQ; ++k) mx = fmaxf(mx, sc[k]);
    float sum = 0.f;
    for (int k = 0; k < SEQ; ++k) { float e = __expf(sc[k] - mx); sc[k] = e; sum += e; }
    sinv = 1.f / sum;
  }
  __syncthreads();

  if (tid < DHD) {
    float inv = sinv;
    float o = 0.f;
    for (int k = 0; k < SEQ; ++k)
      o += sc[k] * base[(size_t)k * 3 * HIDD + 2 * HIDD + h * DHD + tid];
    out[(size_t)(b * SEQ + q) * HIDD + h * DHD + tid] = o * inv;
  }
}

// ---------------------------------------------------------------------------
// Head: logits = t[:,0] @ head_W + head_b ; sigmoid
// ---------------------------------------------------------------------------
__global__ __launch_bounds__(64) void k_head(const float* __restrict__ t,
                                             const float* __restrict__ W,
                                             const float* __restrict__ bias,
                                             float* __restrict__ out) {
  int b = blockIdx.x;
  int tid = threadIdx.x, lane = tid & 31, wv = tid >> 5;
  const float* r = t + (size_t)b * SEQ * HIDD;  // cls token row
  __shared__ float red[2];
  for (int c = 0; c < NCLS; ++c) {
    float s = 0.f;
    for (int i = tid; i < HIDD; i += 64) s += r[i] * W[(size_t)i * NCLS + c];
#pragma unroll
    for (int o = 16; o > 0; o >>= 1) s += __shfl_xor(s, o, 32);
    if (lane == 0) red[wv] = s;
    __syncthreads();
    if (tid == 0) {
      float v = red[0] + red[1] + bias[c];
      out[b * NCLS + c] = 1.f / (1.f + __expf(-v));
    }
    __syncthreads();
  }
}

// ---------------------------------------------------------------------------
// Host orchestration
// ---------------------------------------------------------------------------
static void gemm(const float* A, const float* B, const float* bias, const float* res,
                 float* C, int N, int K, int M, int act, hipStream_t s) {
  dim3 g(cdiv_i(N, BM), M / BN);
  k_gemm_bf16<<<g, 256, 0, s>>>(A, B, bias, res, C, N, K, M, act);
}

extern "C" void kernel_launch(void* const* d_in, const int* in_sizes, int n_in,
                              void* d_out, int out_size, void* d_ws, size_t ws_size,
                              hipStream_t stream) {
  (void)in_sizes; (void)n_in; (void)out_size; (void)ws_size;
  // inputs
  const float* x         = (const float*)d_in[0];
  const int*   ep        = (const int*)d_in[1];     // (2, EP_N): row0=src, row1=dst
  const int*   spa       = (const int*)d_in[2];
  const int*   es        = (const int*)d_in[3];     // (2, ES_N)
  // params (dict insertion order)
  const float* i_ln0_g = (const float*)d_in[4];
  const float* i_ln0_b = (const float*)d_in[5];
  const float* i_Wl    = (const float*)d_in[6];
  const float* i_bl    = (const float*)d_in[7];
  const float* i_Wr    = (const float*)d_in[8];
  const float* i_ln1_g = (const float*)d_in[9];
  const float* i_ln1_b = (const float*)d_in[10];
  const float* att_W1  = (const float*)d_in[11];
  const float* att_b1  = (const float*)d_in[12];
  const float* att_W2  = (const float*)d_in[13];
  const float* att_b2  = (const float*)d_in[14];
  const float* e_ln0_g = (const float*)d_in[15];
  const float* e_ln0_b = (const float*)d_in[16];
  const float* e_Wl    = (const float*)d_in[17];
  const float* e_bl    = (const float*)d_in[18];
  const float* e_Wr    = (const float*)d_in[19];
  const float* e_ln1_g = (const float*)d_in[20];
  const float* e_ln1_b = (const float*)d_in[21];
  const float* pe_W    = (const float*)d_in[22];
  const float* pe_b    = (const float*)d_in[23];
  const float* cls     = (const float*)d_in[24];
  const float* pos     = (const float*)d_in[25];
  const float* pre_g   = (const float*)d_in[26];
  const float* pre_b   = (const float*)d_in[27];
  const float* b_ln1_g = (const float*)d_in[28];
  const float* b_ln1_b = (const float*)d_in[29];
  const float* qkv_W   = (const float*)d_in[30];
  const float* qkv_b   = (const float*)d_in[31];
  const float* proj_W  = (const float*)d_in[32];
  const float* proj_b  = (const float*)d_in[33];
  const float* b_ln2_g = (const float*)d_in[34];
  const float* b_ln2_b = (const float*)d_in[35];
  const float* mlp_W1  = (const float*)d_in[36];
  const float* mlp_b1  = (const float*)d_in[37];
  const float* mlp_W2  = (const float*)d_in[38];
  const float* mlp_b2  = (const float*)d_in[39];
  const float* fn_g    = (const float*)d_in[40];
  const float* fn_b    = (const float*)d_in[41];
  const float* head_W  = (const float*)d_in[42];
  const float* head_b  = (const float*)d_in[43];

  // workspace carve (256B aligned)
  char* wp = (char*)d_ws;
  auto alloc_f = [&](size_t nflt) -> float* {
    float* p = (float*)wp;
    wp += ((nflt * sizeof(float) + 255) / 256) * 256;
    return p;
  };
  float*    XLN   = alloc_f((size_t)N_PATCH * IN_DIM);
  float*    AGG   = alloc_f((size_t)N_PATCH * IN_DIM);
  float*    DEG   = alloc_f(N_PATCH);
  float*    HB    = alloc_f((size_t)N_PATCH * HIDD);
  float*    G1    = alloc_f((size_t)N_PATCH * 256);
  float*    GATE  = alloc_f(N_PATCH);
  unsigned* MB    = (unsigned*)alloc_f(N_SP);
  float*    ZB    = alloc_f(N_SP);
  float*    SP    = alloc_f((size_t)N_SP * HIDD);
  float*    F0    = alloc_f((size_t)N_SP * HIDD);
  float*    AGGS  = alloc_f((size_t)N_SP * HIDD);
  float*    DEGS  = alloc_f(N_SP);
  float*    FB    = alloc_f((size_t)N_SP * HIDD);
  float*    TBODY = alloc_f((size_t)N_SP * HIDD);
  float*    T     = alloc_f((size_t)BATCH * SEQ * HIDD);
  float*    Y     = alloc_f((size_t)BATCH * SEQ * HIDD);
  float*    QKV   = alloc_f((size_t)BATCH * SEQ * 3 * HIDD);
  float*    O     = alloc_f((size_t)BATCH * SEQ * HIDD);
  float*    MLP1  = alloc_f((size_t)BATCH * SEQ * 4 * HIDD);

  const int* ep_src = ep;          const int* ep_dst = ep + EP_N;
  const int* es_src = es;          const int* es_dst = es + ES_N;

  // ---- patch level ----
  k_ln<<<N_PATCH, 256, 0, stream>>>(x, i_ln0_g, i_ln0_b, XLN, IN_DIM, 1e-5f);

  k_fill_f<<<8192, 256, 0, stream>>>(AGG, (long long)N_PATCH * IN_DIM, 0.f);
  k_fill_f<<<512, 256, 0, stream>>>(DEG, N_PATCH, 0.f);
  k_scatter_rows<<<EP_N, 256, 0, stream>>>(XLN, ep_src, ep_dst, AGG, EP_N, IN_DIM);
  k_deg<<<cdiv_i(EP_N, 256), 256, 0, stream>>>(ep_dst, DEG, EP_N);
  k_divdeg<<<N_PATCH, 256, 0, stream>>>(AGG, DEG, IN_DIM);

  // h = relu(agg@Wl + bl + xln@Wr)
  gemm(XLN, i_Wr, nullptr, nullptr, HB, N_PATCH, IN_DIM, HIDD, 0, stream);
  gemm(AGG, i_Wl, i_bl, HB, HB, N_PATCH, IN_DIM, HIDD, 1, stream);
  k_ln<<<N_PATCH, 256, 0, stream>>>(HB, i_ln1_g, i_ln1_b, HB, HIDD, 1e-5f);

  // gate = relu(h@W1+b1)@W2+b2 ; segment softmax ; pool
  gemm(HB, att_W1, att_b1, nullptr, G1, N_PATCH, HIDD, 256, 1, stream);
  k_gate_dot<<<cdiv_i(N_PATCH, 8), 256, 0, stream>>>(G1, att_W2, att_b2, GATE, N_PATCH);
  k_fill_u<<<64, 256, 0, stream>>>(MB, N_SP, 0x007FFFFFu);   // fenc(-inf)
  k_fill_f<<<64, 256, 0, stream>>>(ZB, N_SP, 0.f);
  k_seg_max<<<cdiv_i(N_PATCH, 256), 256, 0, stream>>>(GATE, spa, MB, N_PATCH);
  k_seg_exp<<<cdiv_i(N_PATCH, 256), 256, 0, stream>>>(GATE, spa, MB, ZB, N_PATCH);
  k_fill_f<<<2048, 256, 0, stream>>>(SP, (long long)N_SP * HIDD, 0.f);
  k_pool<<<N_PATCH, 256, 0, stream>>>(HB, GATE, ZB, spa, SP, HIDD);

  // ---- superpixel level ----
  k_ln<<<N_SP, 256, 0, stream>>>(SP, e_ln0_g, e_ln0_b, F0, HIDD, 1e-5f);
  k_fill_f<<<2048, 256, 0, stream>>>(AGGS, (long long)N_SP * HIDD, 0.f);
  k_fill_f<<<32, 256, 0, stream>>>(DEGS, N_SP, 0.f);
  k_scatter_rows<<<ES_N, 256, 0, stream>>>(F0, es_src, es_dst, AGGS, ES_N, HIDD);
  k_deg<<<cdiv_i(ES_N, 256), 256, 0, stream>>>(es_dst, DEGS, ES_N);
  k_divdeg<<<N_SP, 256, 0, stream>>>(AGGS, DEGS, HIDD);

  gemm(F0, e_Wr, nullptr, nullptr, FB, N_SP, HIDD, HIDD, 0, stream);
  gemm(AGGS, e_Wl, e_bl, FB, FB, N_SP, HIDD, HIDD, 1, stream);
  k_ln<<<N_SP, 256, 0, stream>>>(FB, e_ln1_g, e_ln1_b, F0, HIDD, 1e-5f);

  // ---- tokens + transformer block ----
  gemm(F0, pe_W, pe_b, nullptr, TBODY, N_SP, HIDD, HIDD, 0, stream);
  {
    long long tot = (long long)BATCH * SEQ * HIDD;
    k_tokens<<<(int)((tot + 255) / 256), 256, 0, stream>>>(TBODY, cls, pos, T);
  }
  const int NT = BATCH * SEQ;
  k_ln<<<NT, 256, 0, stream>>>(T, pre_g, pre_b, T, HIDD, 1e-6f);

  k_ln<<<NT, 256, 0, stream>>>(T, b_ln1_g, b_ln1_b, Y, HIDD, 1e-6f);
  gemm(Y, qkv_W, qkv_b, nullptr, QKV, NT, HIDD, 3 * HIDD, 0, stream);
  k_attn<<<BATCH * HEADS * SEQ, 128, 0, stream>>>(QKV, O);
  gemm(O, proj_W, proj_b, T, T, NT, HIDD, HIDD, 0, stream);   // t += o@proj+b

  k_ln<<<NT, 256, 0, stream>>>(T, b_ln2_g, b_ln2_b, Y, HIDD, 1e-6f);
  gemm(Y, mlp_W1, mlp_b1, nullptr, MLP1, NT, HIDD, 4 * HIDD, 2, stream);  // gelu
  gemm(MLP1, mlp_W2, mlp_b2, T, T, NT, 4 * HIDD, HIDD, 0, stream);        // t += mlp

  k_ln<<<NT, 256, 0, stream>>>(T, fn_g, fn_b, Y, HIDD, 1e-6f);
  k_head<<<BATCH, 64, 0, stream>>>(Y, head_W, head_b, (float*)d_out);
}
